// MeshGNN_9148280340958
// MI455X (gfx1250) — matrix operand order repaired and checked by reading, not compiled, not run on hardware
//
#include <hip/hip_runtime.h>
#include <hip/hip_bf16.h>

#define HID 128
#define DM  256
#define NB_GRAPH 64
#define LN_EPS 1e-5f

typedef __attribute__((ext_vector_type(16))) _Float16 v16h;
typedef __attribute__((ext_vector_type(8)))  _Float16 v8h;
typedef __attribute__((ext_vector_type(8)))  float    v8f;

// ---------------------------------------------------------------------------
// deg / dinv
// ---------------------------------------------------------------------------
__global__ void k_init_deg(float* __restrict__ deg, int n) {
    int i = blockIdx.x * blockDim.x + threadIdx.x;
    if (i < n) deg[i] = 1.0f;                       // +1 self loop
}
__global__ void k_accum_deg(const int* __restrict__ dst, float* __restrict__ deg, int E) {
    int e = blockIdx.x * blockDim.x + threadIdx.x;
    if (e < E) atomicAdd(&deg[dst[e]], 1.0f);
}
__global__ void k_rsqrt_deg(float* __restrict__ deg, int n) {
    int i = blockIdx.x * blockDim.x + threadIdx.x;
    if (i < n) deg[i] = rsqrtf(deg[i]);             // in-place -> dinv
}

// ---------------------------------------------------------------------------
// input linear: x = vertices @ W_in + b_in  ([N,3]@[3,128])
// writes f32 and f16 copies of x
// ---------------------------------------------------------------------------
__global__ void k_lin_in(const float* __restrict__ V, const float* __restrict__ Win,
                         const float* __restrict__ bin,
                         float* __restrict__ X, _Float16* __restrict__ Xh, int n128) {
    int t = blockIdx.x * blockDim.x + threadIdx.x;
    if (t >= n128) return;
    int i = t >> 7, j = t & 127;
    float y = V[i*3+0]*Win[j] + V[i*3+1]*Win[HID+j] + V[i*3+2]*Win[2*HID+j] + bin[j];
    X[t] = y;
    Xh[t] = (_Float16)y;
}

// ---------------------------------------------------------------------------
// shuffle a [128,128] row-major f32 weight into WMMA B-operand lane layout
// index = ((kt*8 + nt)*32 + lane)*16 + h ; value = W[k0 + (lane>=16?8:0) + (h&7) + (h>=8?16:0)][n0 + (lane&15)]
// ---------------------------------------------------------------------------
__global__ void k_shuffle_W(const float* __restrict__ W, _Float16* __restrict__ Bs) {
    int idx = blockIdx.x * blockDim.x + threadIdx.x;      // 4*8*32*16 = 16384
    if (idx >= 4*8*32*16) return;
    int h    = idx & 15;
    int lane = (idx >> 4) & 31;
    int nt   = (idx >> 9) & 7;
    int kt   = (idx >> 12) & 3;
    int n = nt*16 + (lane & 15);
    int k = kt*32 + ((lane >> 4) << 3) + (h & 7) + ((h >> 3) << 4);
    Bs[idx] = (_Float16)W[k*HID + n];
}

// ---------------------------------------------------------------------------
// H = Xh @ W  via v_wmma_f32_16x16x32_f16 ; one wave per 16x16 tile, 4 K-steps
// A (16-bit 16x32 layout): lane<16 -> row=lane, K={k0..k0+7, k0+16..k0+23}
//                          lane>=16 -> row=lane-16, K={k0+8..k0+15, k0+24..k0+31}
// ---------------------------------------------------------------------------
__global__ void k_gemm_wmma(const _Float16* __restrict__ A, const _Float16* __restrict__ Bs,
                            float* __restrict__ H, int n_mtiles) {
    int wave = (int)((blockIdx.x * (size_t)blockDim.x + threadIdx.x) >> 5);
    int lane = threadIdx.x & 31;
    int mt = wave >> 3;          // 8 n-tiles per m-tile (HID/16)
    int nt = wave & 7;
    if (mt >= n_mtiles) return;
    int row0 = mt << 4;
    int r    = row0 + (lane & 15);
    int koff = (lane >> 4) << 3;                 // +8 halves for upper lane group
    v8f acc = {};
#pragma unroll
    for (int kt = 0; kt < 4; ++kt) {
        int kbase = kt*32 + koff;
        const _Float16* ap = A + (size_t)r*HID + kbase;
        v8h lo = *(const v8h*)(ap);
        v8h hi = *(const v8h*)(ap + 16);
        v16h a;
#pragma unroll
        for (int i = 0; i < 8; ++i) { a[i] = lo[i]; a[i+8] = hi[i]; }
        v16h b = *(const v16h*)(Bs + (((kt*8 + nt)*32 + lane) << 4));
        acc = __builtin_amdgcn_wmma_f32_16x16x32_f16(false, a, false, b,
                                                     (short)0, acc, false, false);
    }
    // C layout: VGPR v -> lanes 0-15: M=v ; lanes 16-31: M=8+v ; N = lane&15
    int col   = nt*16 + (lane & 15);
    int rbase = row0 + ((lane >> 4) << 3);
#pragma unroll
    for (int v = 0; v < 8; ++v)
        H[(size_t)(rbase + v)*HID + col] = acc[v];
}

// ---------------------------------------------------------------------------
// agg = h * dinv^2 + bias   (self-loop term + bias folded in)
// ---------------------------------------------------------------------------
__global__ void k_init_agg(const float* __restrict__ H, const float* __restrict__ dinv,
                           const float* __restrict__ bias, float* __restrict__ AGG, int n128) {
    int t = blockIdx.x * blockDim.x + threadIdx.x;
    if (t >= n128) return;
    int i = t >> 7, j = t & 127;
    float di = dinv[i];
    AGG[t] = H[t]*di*di + bias[j];
}

// ---------------------------------------------------------------------------
// edge scatter: agg[dst] += h[src] * dinv[src]*dinv[dst] ; 32 lanes/edge, float4/lane
// ---------------------------------------------------------------------------
__global__ void k_scatter(const int* __restrict__ src, const int* __restrict__ dst,
                          const float* __restrict__ dinv, const float* __restrict__ H,
                          float* __restrict__ AGG, int E) {
    size_t t = blockIdx.x * (size_t)blockDim.x + threadIdx.x;
    int e = (int)(t >> 5);
    if (e >= E) return;
    int c = ((int)t & 31) << 2;
    int s = src[e], d = dst[e];
    float w = dinv[s] * dinv[d];
    const float4 hv = *(const float4*)(H + (size_t)s*HID + c);
    float* ap = AGG + (size_t)d*HID + c;
    atomicAdd(ap + 0, hv.x * w);
    atomicAdd(ap + 1, hv.y * w);
    atomicAdd(ap + 2, hv.z * w);
    atomicAdd(ap + 3, hv.w * w);
}

// ---------------------------------------------------------------------------
// x = relu(agg) ; write f32 and f16 copies
// ---------------------------------------------------------------------------
__global__ void k_relu(const float* __restrict__ AGG, float* __restrict__ X,
                       _Float16* __restrict__ Xh, int n128) {
    int t = blockIdx.x * blockDim.x + threadIdx.x;
    if (t >= n128) return;
    float v = fmaxf(AGG[t], 0.0f);
    X[t] = v;
    Xh[t] = (_Float16)v;
}

// ---------------------------------------------------------------------------
// pooling
// ---------------------------------------------------------------------------
__global__ void k_zero_pool(float* __restrict__ pool, float* __restrict__ cnt) {
    int t = blockIdx.x * blockDim.x + threadIdx.x;
    if (t < NB_GRAPH*HID) pool[t] = 0.0f;
    if (t < NB_GRAPH)     cnt[t]  = 0.0f;
}
__global__ void k_pool_accum(const float* __restrict__ X, const int* __restrict__ batch,
                             float* __restrict__ pool, int n) {
    size_t t = blockIdx.x * (size_t)blockDim.x + threadIdx.x;
    int i = (int)(t >> 5);
    if (i >= n) return;
    int c = ((int)t & 31) << 2;
    int b = batch[i];
    const float4 xv = *(const float4*)(X + (size_t)i*HID + c);
    float* p = pool + b*HID + c;
    atomicAdd(p + 0, xv.x);
    atomicAdd(p + 1, xv.y);
    atomicAdd(p + 2, xv.z);
    atomicAdd(p + 3, xv.w);
}
__global__ void k_cnt_accum(const int* __restrict__ batch, float* __restrict__ cnt, int n) {
    int i = blockIdx.x * blockDim.x + threadIdx.x;
    if (i < n) atomicAdd(&cnt[batch[i]], 1.0f);
}

// ---------------------------------------------------------------------------
// y = (pool/cnt) @ W_out + b_out ; LayerNorm over DM ; one block per graph row
// ---------------------------------------------------------------------------
__global__ void k_out_ln(const float* __restrict__ pool, const float* __restrict__ cnt,
                         const float* __restrict__ Wout, const float* __restrict__ bout,
                         const float* __restrict__ gamma, const float* __restrict__ beta,
                         float* __restrict__ out) {
    __shared__ float p[HID];
    __shared__ float red[DM];
    int b = blockIdx.x, j = threadIdx.x;
    if (j < HID) p[j] = pool[b*HID + j] / fmaxf(cnt[b], 1.0f);
    __syncthreads();
    float y = bout[j];
#pragma unroll 8
    for (int k = 0; k < HID; ++k) y += p[k] * Wout[k*DM + j];
    // mean
    red[j] = y; __syncthreads();
    for (int s = DM/2; s > 0; s >>= 1) { if (j < s) red[j] += red[j + s]; __syncthreads(); }
    float mu = red[0] / (float)DM; __syncthreads();
    // variance (population)
    float d = y - mu;
    red[j] = d*d; __syncthreads();
    for (int s = DM/2; s > 0; s >>= 1) { if (j < s) red[j] += red[j + s]; __syncthreads(); }
    float var = red[0] / (float)DM;
    out[b*DM + j] = d * rsqrtf(var + LN_EPS) * gamma[j] + beta[j];
}

// ---------------------------------------------------------------------------
extern "C" void kernel_launch(void* const* d_in, const int* in_sizes, int n_in,
                              void* d_out, int out_size, void* d_ws, size_t ws_size,
                              hipStream_t stream) {
    const float* vertices = (const float*)d_in[0];
    const int*   eidx     = (const int*)  d_in[1];
    const int*   batch    = (const int*)  d_in[2];
    const float* W_in     = (const float*)d_in[3];
    const float* b_in     = (const float*)d_in[4];
    const float* Ws[3]    = {(const float*)d_in[5], (const float*)d_in[7], (const float*)d_in[9]};
    const float* bs[3]    = {(const float*)d_in[6], (const float*)d_in[8], (const float*)d_in[10]};
    const float* W_out    = (const float*)d_in[11];
    const float* b_out    = (const float*)d_in[12];
    const float* gamma    = (const float*)d_in[13];
    const float* beta     = (const float*)d_in[14];
    float* out = (float*)d_out;

    const int N = in_sizes[0] / 3;
    const int E = in_sizes[1] / 2;
    const int* src = eidx;
    const int* dst = eidx + E;
    const int n128 = N * HID;

    // ---- workspace carve ----
    char* ws = (char*)d_ws;
    size_t off = 0;
    auto carve = [&](size_t bytes) -> char* {
        char* p = ws + off;
        off = (off + bytes + 255) & ~(size_t)255;
        return p;
    };
    float*    X    = (float*)   carve((size_t)n128 * 4);
    _Float16* Xh   = (_Float16*)carve((size_t)n128 * 2);
    float*    H    = (float*)   carve((size_t)n128 * 4);
    float*    AGG  = (float*)   carve((size_t)n128 * 4);
    float*    DINV = (float*)   carve((size_t)N * 4);
    _Float16* WSH  = (_Float16*)carve((size_t)3 * 4*8*32*16 * 2);
    float*    POOL = (float*)   carve((size_t)NB_GRAPH * HID * 4);
    float*    CNT  = (float*)   carve((size_t)NB_GRAPH * 4);
    (void)ws_size;

    const int T = 256;
    auto blk = [](long long n, int t) { return (unsigned)((n + t - 1) / t); };

    // degree / dinv (shared by all 3 layers)
    k_init_deg  <<<blk(N, T), T, 0, stream>>>(DINV, N);
    k_accum_deg <<<blk(E, T), T, 0, stream>>>(dst, DINV, E);
    k_rsqrt_deg <<<blk(N, T), T, 0, stream>>>(DINV, N);

    // weight shuffles into WMMA B layout
    for (int l = 0; l < 3; ++l)
        k_shuffle_W<<<blk(4*8*32*16, T), T, 0, stream>>>(Ws[l], WSH + (size_t)l*4*8*32*16);

    // input linear
    k_lin_in<<<blk(n128, T), T, 0, stream>>>(vertices, W_in, b_in, X, Xh, n128);

    // 3 GCN layers
    const int n_mtiles = N / 16;                       // 6250
    const long long gemm_threads = (long long)n_mtiles * 8 * 32;
    for (int l = 0; l < 3; ++l) {
        k_gemm_wmma<<<blk(gemm_threads, T), T, 0, stream>>>(Xh, WSH + (size_t)l*4*8*32*16, H, n_mtiles);
        k_init_agg <<<blk(n128, T), T, 0, stream>>>(H, DINV, bs[l], AGG, n128);
        k_scatter  <<<blk((long long)E * 32, T), T, 0, stream>>>(src, dst, DINV, H, AGG, E);
        k_relu     <<<blk(n128, T), T, 0, stream>>>(AGG, X, Xh, n128);
    }

    // global mean pool + output linear + layernorm
    k_zero_pool <<<blk(NB_GRAPH*HID, T), T, 0, stream>>>(POOL, CNT);
    k_pool_accum<<<blk((long long)N * 32, T), T, 0, stream>>>(X, batch, POOL, N);
    k_cnt_accum <<<blk(N, T), T, 0, stream>>>(batch, CNT, N);
    k_out_ln    <<<NB_GRAPH, DM, 0, stream>>>(POOL, CNT, W_out, b_out, gamma, beta, out);
}
